// TransformerShardA_35476429865973
// MI455X (gfx1250) — matrix operand order repaired and checked by reading, not compiled, not run on hardware
//
#include <hip/hip_runtime.h>

typedef __attribute__((ext_vector_type(16))) __bf16       v16bf;
typedef __attribute__((ext_vector_type(8)))  float        v8f;
typedef __attribute__((ext_vector_type(8)))  unsigned int v8u;
typedef __attribute__((ext_vector_type(4)))  unsigned int v4u;
typedef __attribute__((ext_vector_type(4)))  int          v4i;
typedef __attribute__((ext_vector_type(8)))  int          v8i;

#define S_LEN 2048
#define DMODEL 768
#define FFND 3072
#define NHEAD 12
#define HDIM 64

#if __has_builtin(__builtin_amdgcn_tensor_load_to_lds)
#define HAVE_TDM 1
#else
#define HAVE_TDM 0
#endif

__device__ __forceinline__ unsigned short f2bf(float f) {
    unsigned int u = __builtin_bit_cast(unsigned int, f);
    u += 0x7FFFu + ((u >> 16) & 1u);          // round-to-nearest-even
    return (unsigned short)(u >> 16);
}

#if HAVE_TDM
// 2D TDM load of bf16 tile [tile_d1 rows x tile_d0 elems] into LDS with row
// padding.  pad_interval/pad_amount are the encoded D# fields (ISA 8.4).
__device__ __forceinline__ void tdm_load_2d(unsigned int lds_off,
                                            const void* gaddr,
                                            unsigned int tensor_d0,
                                            unsigned int tensor_d1,
                                            unsigned int tile_d0,
                                            unsigned int tile_d1,
                                            unsigned int stride_d0,
                                            unsigned int pad_interval,
                                            unsigned int pad_amount) {
    unsigned long long ga = (unsigned long long)(size_t)gaddr;
    v4u g0;
    g0[0] = 1u;                                            // count=1 (user D#)
    g0[1] = lds_off;                                       // lds_addr
    g0[2] = (unsigned int)(ga & 0xFFFFFFFFu);              // global_addr lo
    g0[3] = (unsigned int)((ga >> 32) & 0x01FFFFFFu) | 0x80000000u; // hi|type=2
    v8i g1;
    g1[0] = (int)((1u << 16)                               // data_size = 2B
                | (1u << 20)                               // pad_enable
                | (pad_interval << 22) | (pad_amount << 25));
    g1[1] = (int)((tensor_d0 & 0xFFFFu) << 16);            // tensor_dim0 lo
    g1[2] = (int)(((tensor_d0 >> 16) & 0xFFFFu) | ((tensor_d1 & 0xFFFFu) << 16));
    g1[3] = (int)(((tensor_d1 >> 16) & 0xFFFFu) | ((tile_d0 & 0xFFFFu) << 16));
    g1[4] = (int)(tile_d1 & 0xFFFFu);                      // tile_dim1 (dim2=0)
    g1[5] = (int)stride_d0;                                // dim0 stride lo
    g1[6] = 0;
    g1[7] = 0;
    v4i z4 = {0, 0, 0, 0};
#if __clang_major__ >= 23
    v8i z8 = {0, 0, 0, 0, 0, 0, 0, 0};
    __builtin_amdgcn_tensor_load_to_lds(g0, g1, z4, z4, z8, 0);
#else
    __builtin_amdgcn_tensor_load_to_lds(g0, g1, z4, z4, 0);
#endif
}
#endif

// ---------------------------------------------------------------- embed ----
__global__ void embed_kernel(const int* __restrict__ idx,
                             const float* __restrict__ temb,
                             const float* __restrict__ pe,
                             float* __restrict__ x) {
    int i = blockIdx.x * blockDim.x + threadIdx.x;      // S*D threads
    int s = i / DMODEL, d = i % DMODEL;
    x[i] = temb[(size_t)idx[s] * DMODEL + d] + pe[i];
}

// ------------------------------------------------------------ layernorm ----
__global__ void ln_bf16_kernel(const float* __restrict__ x,
                               const float* __restrict__ sc,
                               const float* __restrict__ bi,
                               unsigned short* __restrict__ out) {
    __shared__ float red[256];
    int row = blockIdx.x, t = threadIdx.x;
    const float* xr = x + (size_t)row * DMODEL;
    float v0 = xr[t], v1 = xr[t + 256], v2 = xr[t + 512];
    red[t] = v0 + v1 + v2;
    __syncthreads();
    for (int off = 128; off > 0; off >>= 1) {
        if (t < off) red[t] += red[t + off];
        __syncthreads();
    }
    float mean = red[0] * (1.0f / DMODEL);
    __syncthreads();
    float d0 = v0 - mean, d1 = v1 - mean, d2 = v2 - mean;
    red[t] = d0 * d0 + d1 * d1 + d2 * d2;
    __syncthreads();
    for (int off = 128; off > 0; off >>= 1) {
        if (t < off) red[t] += red[t + off];
        __syncthreads();
    }
    float rstd = rsqrtf(red[0] * (1.0f / DMODEL) + 1e-5f);
    unsigned short* o = out + (size_t)row * DMODEL;
    o[t]       = f2bf(d0 * rstd * sc[t]       + bi[t]);
    o[t + 256] = f2bf(d1 * rstd * sc[t + 256] + bi[t + 256]);
    o[t + 512] = f2bf(d2 * rstd * sc[t + 512] + bi[t + 512]);
}

// ------------------------------------------------------ generic WMMA GEMM --
// C[M,N] = act(A_bf16[M,K] @ B_f32->bf16[K,N] + bias) (+resid), out f32/bf16.
// Block = 256 threads (8 waves), tile 64x128, K stepped by 32.
// A tile staged by TDM (8 rows per wave, padded rows); B via VALU convert.
__global__ void gemm_bf16_kernel(const unsigned short* __restrict__ A,
                                 const float* __restrict__ B,
                                 const float* __restrict__ bias,
                                 const float* __restrict__ resid,
                                 float* __restrict__ Cf,
                                 unsigned short* __restrict__ Cbf,
                                 int M, int N, int K, int act) {
    __shared__ unsigned int LA[64 * 18];    // [m][k-pair], 16 dwords + 2 pad
    __shared__ unsigned int LB[128 * 18];   // [n][k-pair]
    int t = threadIdx.x;
    int lane = t & 31, wave = t >> 5, hi = lane >> 4;
    int mb = blockIdx.y * 64, nb = blockIdx.x * 128;
    int m0 = (wave & 3) * 16, n0 = (wave >> 2) * 64;

    v8f zero = {0, 0, 0, 0, 0, 0, 0, 0};
    v8f acc[4];
#pragma unroll
    for (int i = 0; i < 4; ++i) acc[i] = zero;

    const unsigned int* Au = (const unsigned int*)A;
    int Kd = K >> 1;
    (void)Au; (void)Kd;

    for (int kb = 0; kb < K; kb += 32) {
#if HAVE_TDM
        // each wave DMAs its own 8-row slice of the 64x32 A tile
        tdm_load_2d((unsigned int)(size_t)&LA[wave * 8 * 18],
                    A + ((size_t)(mb + wave * 8) * K + kb),
                    (unsigned int)K, (unsigned int)M,
                    32u, 8u, (unsigned int)K,
                    3u /*16 dwords*/, 1u /*2 dwords*/);
#else
#pragma unroll
        for (int i = 0; i < 4; ++i) {
            int dw = t * 4 + i;
            int r = dw >> 4, c = dw & 15;
            LA[r * 18 + c] = Au[(size_t)(mb + r) * Kd + (kb >> 1) + c];
        }
#endif
        // B tile: convert f32 weights -> bf16 pairs, store n-major
        {
            int n = t & 127;
            int kk0 = (t >> 7) * 16;
            if (kb + 32 < K)
                __builtin_prefetch(&B[(size_t)(kb + 32 + kk0) * N + nb + n], 0, 1);
#pragma unroll
            for (int j = 0; j < 16; j += 2) {
                float f0 = B[(size_t)(kb + kk0 + j)     * N + nb + n];
                float f1 = B[(size_t)(kb + kk0 + j + 1) * N + nb + n];
                LB[n * 18 + ((kk0 + j) >> 1)] =
                    (unsigned int)f2bf(f0) | ((unsigned int)f2bf(f1) << 16);
            }
        }
#if HAVE_TDM
        __builtin_amdgcn_s_wait_tensorcnt(0);
#endif
        __syncthreads();

        v8u au;
        int ml = m0 + (lane & 15);
#pragma unroll
        for (int v = 0; v < 8; ++v) {
            int kpA = (v & 3) + (hi ? 4 : 0) + (v < 4 ? 0 : 8);
            au[v] = LA[ml * 18 + kpA];
        }
        v16bf a = __builtin_bit_cast(v16bf, au);
#pragma unroll
        for (int t4 = 0; t4 < 4; ++t4) {
            v8u bu;
            int nl = n0 + t4 * 16 + (lane & 15);
#pragma unroll
            for (int v = 0; v < 8; ++v)
                bu[v] = LB[nl * 18 + (v & 3) + (hi ? 4 : 0) + (v < 4 ? 0 : 8)];
            acc[t4] = __builtin_amdgcn_wmma_f32_16x16x32_bf16(
                false, a, false, __builtin_bit_cast(v16bf, bu),
                (short)0, acc[t4], false, false);
        }
        __syncthreads();
    }

#pragma unroll
    for (int r = 0; r < 8; ++r) {
        int m = mb + m0 + r + hi * 8;
#pragma unroll
        for (int t4 = 0; t4 < 4; ++t4) {
            int ng = nb + n0 + t4 * 16 + (lane & 15);
            float v = acc[t4][r] + bias[ng];
            if (act == 1)                          // exact GELU
                v = 0.5f * v * (1.0f + erff(v * 0.70710678f));
            if (resid) v += resid[(size_t)m * N + ng];
            if (Cf)  Cf[(size_t)m * N + ng]  = v;
            if (Cbf) Cbf[(size_t)m * N + ng] = f2bf(v);
        }
    }
}

// ----------------------------------------------------------- QKV repack ----
__global__ void repack_kernel(const float* __restrict__ qkv,
                              unsigned short* __restrict__ Qb,
                              unsigned short* __restrict__ Kb,
                              unsigned short* __restrict__ Vb) {
    int i = blockIdx.x * blockDim.x + threadIdx.x;      // S*D threads
    int s = i / DMODEL, c = i % DMODEL;
    int h = c >> 6, d = c & 63;
    size_t dst = ((size_t)h * S_LEN + s) * HDIM + d;
    size_t base = (size_t)s * (3 * DMODEL);
    Qb[dst] = f2bf(qkv[base + c]);
    Kb[dst] = f2bf(qkv[base + DMODEL + c]);
    Vb[dst] = f2bf(qkv[base + 2 * DMODEL + c]);
}

// ------------------------------------------------------- flash attention ---
// Block = 128 threads (4 waves), 64 query rows per block, 32-key chunks.
__global__ void flash_attn_kernel(const unsigned short* __restrict__ Qb,
                                  const unsigned short* __restrict__ Kb,
                                  const unsigned short* __restrict__ Vb,
                                  unsigned short* __restrict__ Obf) {
    __shared__ unsigned int LK[32 * 34];        // [key][d-pair]  32 dw + 2 pad
    __shared__ unsigned int LV[64 * 18];        // [d][j-pair]    (V transposed)
    __shared__ unsigned int LP[4 * 16 * 18];    // per-wave P tile [row][j-pair]

    int t = threadIdx.x, lane = t & 31, wave = t >> 5, hi = lane >> 4;
    int qb0 = blockIdx.x * 64, h = blockIdx.y;
    const unsigned int* Qu = (const unsigned int*)Qb;
    const unsigned int* Ku = (const unsigned int*)Kb;
    (void)Ku;

    int mrow = qb0 + wave * 16 + (lane & 15);
    size_t qrow = ((size_t)h * S_LEN + mrow) * 32;   // 32 dwords / row
    v8u q0u, q1u;
#pragma unroll
    for (int v = 0; v < 8; ++v) {
        int kp = (v & 3) + (hi ? 4 : 0) + (v < 4 ? 0 : 8);
        q0u[v] = Qu[qrow + kp];
        q1u[v] = Qu[qrow + 16 + kp];
    }
    v16bf qa0 = __builtin_bit_cast(v16bf, q0u);
    v16bf qa1 = __builtin_bit_cast(v16bf, q1u);

    v8f zero = {0, 0, 0, 0, 0, 0, 0, 0};
    v8f acc[4];
    float rmax[8], rsum[8];
#pragma unroll
    for (int i = 0; i < 4; ++i) acc[i] = zero;
#pragma unroll
    for (int r = 0; r < 8; ++r) { rmax[r] = -3.0e38f; rsum[r] = 0.0f; }

    for (int kb = 0; kb < qb0 + 64; kb += 32) {
#if HAVE_TDM
        // each wave DMAs 8 keys (rows of 64 bf16) into LK with +2dw row pad
        tdm_load_2d((unsigned int)(size_t)&LK[wave * 8 * 34],
                    Kb + ((size_t)h * S_LEN + kb + wave * 8) * HDIM,
                    64u, (unsigned int)S_LEN,
                    64u, 8u, 64u,
                    4u /*32 dwords*/, 1u /*2 dwords*/);
#else
#pragma unroll
        for (int i = 0; i < 8; ++i) {
            int dw = t * 8 + i;
            int j = dw >> 5, c = dw & 31;
            LK[j * 34 + c] = Ku[((size_t)h * S_LEN + kb + j) * 32 + c];
        }
#endif
        // V chunk transposed: LV[d][j]
        {
            int d = t & 63, jb = (t >> 6) * 16;
#pragma unroll
            for (int j = 0; j < 16; j += 2) {
                unsigned short a0 = Vb[((size_t)h * S_LEN + kb + jb + j)     * HDIM + d];
                unsigned short a1 = Vb[((size_t)h * S_LEN + kb + jb + j + 1) * HDIM + d];
                LV[d * 18 + ((jb + j) >> 1)] =
                    (unsigned int)a0 | ((unsigned int)a1 << 16);
            }
        }
#if HAVE_TDM
        __builtin_amdgcn_s_wait_tensorcnt(0);
#endif
        __syncthreads();

        // scores = Q K^T : two n-tiles x two k-steps
        v8f s0 = zero, s1 = zero;
        v8u b0u, b1u;
#pragma unroll
        for (int v = 0; v < 8; ++v) {
            int kp = (v & 3) + (hi ? 4 : 0) + (v < 4 ? 0 : 8);
            b0u[v] = LK[(lane & 15) * 34 + kp];
            b1u[v] = LK[((lane & 15) + 16) * 34 + kp];
        }
        s0 = __builtin_amdgcn_wmma_f32_16x16x32_bf16(false, qa0, false,
                 __builtin_bit_cast(v16bf, b0u), (short)0, s0, false, false);
        s1 = __builtin_amdgcn_wmma_f32_16x16x32_bf16(false, qa0, false,
                 __builtin_bit_cast(v16bf, b1u), (short)0, s1, false, false);
#pragma unroll
        for (int v = 0; v < 8; ++v) {
            int kp = 16 + (v & 3) + (hi ? 4 : 0) + (v < 4 ? 0 : 8);
            b0u[v] = LK[(lane & 15) * 34 + kp];
            b1u[v] = LK[((lane & 15) + 16) * 34 + kp];
        }
        s0 = __builtin_amdgcn_wmma_f32_16x16x32_bf16(false, qa1, false,
                 __builtin_bit_cast(v16bf, b0u), (short)0, s0, false, false);
        s1 = __builtin_amdgcn_wmma_f32_16x16x32_bf16(false, qa1, false,
                 __builtin_bit_cast(v16bf, b1u), (short)0, s1, false, false);

        // scale + causal mask + online softmax
        int col0 = kb + (lane & 15);
        int col1 = kb + 16 + (lane & 15);
        int rowg = qb0 + wave * 16 + hi * 8;
        unsigned short* LPs = (unsigned short*)LP;
#pragma unroll
        for (int r = 0; r < 8; ++r) {
            int row = rowg + r;
            float x0 = s0[r] * 0.125f; if (col0 > row) x0 = -1e9f;
            float x1 = s1[r] * 0.125f; if (col1 > row) x1 = -1e9f;
            float mx = fmaxf(x0, x1);
            mx = fmaxf(mx, __shfl_xor(mx, 1, 32));
            mx = fmaxf(mx, __shfl_xor(mx, 2, 32));
            mx = fmaxf(mx, __shfl_xor(mx, 4, 32));
            mx = fmaxf(mx, __shfl_xor(mx, 8, 32));
            float nm = fmaxf(rmax[r], mx);
            float cs = __expf(rmax[r] - nm);
            rmax[r] = nm;
            float p0 = __expf(x0 - nm);
            float p1 = __expf(x1 - nm);
            float ps = p0 + p1;
            ps += __shfl_xor(ps, 1, 32);
            ps += __shfl_xor(ps, 2, 32);
            ps += __shfl_xor(ps, 4, 32);
            ps += __shfl_xor(ps, 8, 32);
            rsum[r] = rsum[r] * cs + ps;
            acc[0][r] *= cs; acc[1][r] *= cs; acc[2][r] *= cs; acc[3][r] *= cs;
            int rl = r + hi * 8;
            LPs[(wave * 16 + rl) * 36 + (lane & 15)]      = f2bf(p0);
            LPs[(wave * 16 + rl) * 36 + 16 + (lane & 15)] = f2bf(p1);
        }
        __syncthreads();

        // acc += P @ V
        v8u pu;
#pragma unroll
        for (int v = 0; v < 8; ++v) {
            int kp = (v & 3) + (hi ? 4 : 0) + (v < 4 ? 0 : 8);
            pu[v] = LP[(wave * 16 + (lane & 15)) * 18 + kp];
        }
        v16bf pa = __builtin_bit_cast(v16bf, pu);
#pragma unroll
        for (int t4 = 0; t4 < 4; ++t4) {
            v8u bvu;
#pragma unroll
            for (int v = 0; v < 8; ++v)
                bvu[v] = LV[(t4 * 16 + (lane & 15)) * 18 + (v & 3) + (hi ? 4 : 0) + (v < 4 ? 0 : 8)];
            acc[t4] = __builtin_amdgcn_wmma_f32_16x16x32_bf16(false, pa, false,
                          __builtin_bit_cast(v16bf, bvu), (short)0, acc[t4],
                          false, false);
        }
        __syncthreads();
    }

    // normalize and scatter to O[s, h*64+d]
#pragma unroll
    for (int r = 0; r < 8; ++r) {
        float inv = 1.0f / rsum[r];
        int row = qb0 + wave * 16 + hi * 8 + r;
#pragma unroll
        for (int t4 = 0; t4 < 4; ++t4) {
            int d = t4 * 16 + (lane & 15);
            Obf[(size_t)row * DMODEL + h * HDIM + d] = f2bf(acc[t4][r] * inv);
        }
    }
}

// ------------------------------------------------------------- launcher ----
extern "C" void kernel_launch(void* const* d_in, const int* in_sizes, int n_in,
                              void* d_out, int out_size, void* d_ws, size_t ws_size,
                              hipStream_t stream) {
    (void)in_sizes; (void)n_in; (void)out_size; (void)ws_size;
    const int S = S_LEN, D = DMODEL, F = FFND, H = NHEAD, L = 2;

    const int*   idx   = (const int*)  d_in[0];
    const float* temb  = (const float*)d_in[1];
    const float* pe    = (const float*)d_in[2];
    const float* qkvw  = (const float*)d_in[3];
    const float* qkvb  = (const float*)d_in[4];
    const float* outw  = (const float*)d_in[5];
    const float* outb  = (const float*)d_in[6];
    const float* ln1s  = (const float*)d_in[7];
    const float* ln1b  = (const float*)d_in[8];
    const float* ln2s  = (const float*)d_in[9];
    const float* ln2b  = (const float*)d_in[10];
    const float* w1    = (const float*)d_in[11];
    const float* b1    = (const float*)d_in[12];
    const float* w2    = (const float*)d_in[13];
    const float* b2    = (const float*)d_in[14];

    char* ws = (char*)d_ws;
    size_t off = 0;
    auto alloc = [&](size_t bytes) -> void* {
        void* p = ws + off;
        off += (bytes + 255) & ~(size_t)255;
        return p;
    };
    float*          xa   = (float*)alloc((size_t)S * D * 4);
    float*          xb   = (float*)alloc((size_t)S * D * 4);
    unsigned short* hbf  = (unsigned short*)alloc((size_t)S * D * 2);
    float*          qkvf = (float*)alloc((size_t)S * 3 * D * 4);
    unsigned short* Qb   = (unsigned short*)alloc((size_t)S * D * 2);
    unsigned short* Kb   = (unsigned short*)alloc((size_t)S * D * 2);
    unsigned short* Vb   = (unsigned short*)alloc((size_t)S * D * 2);
    unsigned short* Obf  = (unsigned short*)alloc((size_t)S * D * 2);
    unsigned short* f1   = (unsigned short*)alloc((size_t)S * F * 2);

    embed_kernel<<<(S * D) / 256, 256, 0, stream>>>(idx, temb, pe, xa);

    float* xcur = xa;
    for (int l = 0; l < L; ++l) {
        ln_bf16_kernel<<<S, 256, 0, stream>>>(xcur, ln1s + l * D, ln1b + l * D, hbf);

        gemm_bf16_kernel<<<dim3((3 * D) / 128, S / 64), 256, 0, stream>>>(
            hbf, qkvw + (size_t)l * D * 3 * D, qkvb + (size_t)l * 3 * D,
            nullptr, qkvf, nullptr, S, 3 * D, D, 0);

        repack_kernel<<<(S * D) / 256, 256, 0, stream>>>(qkvf, Qb, Kb, Vb);

        flash_attn_kernel<<<dim3(S / 64, H), 128, 0, stream>>>(Qb, Kb, Vb, Obf);

        gemm_bf16_kernel<<<dim3(D / 128, S / 64), 256, 0, stream>>>(
            Obf, outw + (size_t)l * D * D, outb + (size_t)l * D,
            xcur, xb, nullptr, S, D, D, 0);

        ln_bf16_kernel<<<S, 256, 0, stream>>>(xb, ln2s + l * D, ln2b + l * D, hbf);

        gemm_bf16_kernel<<<dim3(F / 128, S / 64), 256, 0, stream>>>(
            hbf, w1 + (size_t)l * D * F, b1 + (size_t)l * F,
            nullptr, nullptr, f1, S, F, D, 1);

        float* xout = (l == L - 1) ? (float*)d_out : xa;
        gemm_bf16_kernel<<<dim3(D / 128, S / 64), 256, 0, stream>>>(
            f1, w2 + (size_t)l * F * D, b2 + (size_t)l * D,
            xb, xout, nullptr, S, D, F, 0);
        xcur = xout;
    }
}
